// LongformerSelfAttentionForBart_83803401880363
// MI455X (gfx1250) — compile-verified, matrix operand order checked by reading, not run on hardware
//
#include <hip/hip_runtime.h>

// ---------------- problem constants (fixed by setup_inputs) ----------------
constexpr int Bc = 2, Sc = 4096, Ec = 1024, Hc = 16, Dc = 64, Wc = 256, Gc = 64;
constexpr int NCc  = Sc / Wc;      // 16 chunks
constexpr int WK   = 3 * Wc;       // 768 window columns
constexpr int NCOL = Gc + WK;      // 832 total score columns
constexpr long BSr = (long)Bc * Sc;  // 8192 rows

// ---------------- vector types for WMMA ----------------
typedef __attribute__((ext_vector_type(16))) __bf16 v16bf;
typedef __attribute__((ext_vector_type(8)))  __bf16 v8bf;
typedef __attribute__((ext_vector_type(8)))  float  v8f;

union FragU { v16bf v; v8bf h[2]; };

static __device__ inline v8f vzero8() {
  v8f z = {0.f,0.f,0.f,0.f,0.f,0.f,0.f,0.f};
  return z;
}

static __device__ inline int imin(int a, int b) { return a < b ? a : b; }
static __device__ inline int iclampr(int x) { // clamp row into [0, Sc-1]
  x = x < 0 ? 0 : x;
  return x > (Sc - 1) ? (Sc - 1) : x;
}

// Load a 16x32 bf16 fragment (A layout, or B from N-major storage).
// lane&15 selects the row; lane>>4 selects K-half {0..7,16..23} vs {8..15,24..31}.
static __device__ inline v16bf load_frag(const __bf16* base, int ld, int lane) {
  int r  = lane & 15;
  int kb = (lane >> 4) << 3;
  const __bf16* p = base + (long)r * ld + kb;
  FragU f;
  f.h[0] = *(const v8bf*)(p);
  f.h[1] = *(const v8bf*)(p + 16);
  return f.v;
}

static __device__ inline v8f wmma_bf16(v16bf a, v16bf b, v8f c) {
  return __builtin_amdgcn_wmma_f32_16x16x32_bf16(
      false, a, false, b, (short)0, c, false, false);
}

// ---------------- Tensor Data Mover (gfx1250) ----------------
#if defined(__gfx1250__)
#if __has_builtin(__builtin_amdgcn_tensor_load_to_lds) && \
    __has_builtin(__builtin_amdgcn_s_wait_tensorcnt)
#define USE_TDM 1
#endif
#endif

#ifdef USE_TDM
typedef __attribute__((ext_vector_type(4))) unsigned tdm_v4u;
typedef __attribute__((ext_vector_type(8))) int      tdm_v8i;
typedef __attribute__((ext_vector_type(4))) int      tdm_v4i;

static __device__ inline unsigned lds_addr_of(const void* p) {
  return (unsigned)(unsigned long long)
      (const __attribute__((address_space(3))) void*)p;
}

// 2-D tile load: tile_rows x tile_cols bf16 elements, row stride in elements.
// D# packing per CDNA5 ISA 8.3/8.4: group0 = {count|flags, lds_addr, gaddr_lo,
// gaddr_hi|type=2}; group1 = {data_size, tensor_dim0/1, tile_dim0/1/2, strides}.
// This toolchain exposes the 6-arg builtin: (v4u, v8i, v4i, v4i, v8i, cpol).
static __device__ inline void tdm_load_tile_bf16(
    const __bf16* gsrc, unsigned lds_byte_addr,
    unsigned tile_cols, unsigned tile_rows,
    unsigned row_stride_elems, unsigned tensor_cols, unsigned tensor_rows) {
  unsigned long long ga = (unsigned long long)gsrc;
  tdm_v4u g0;
  g0.x = 1u;                                                 // count=1 (valid)
  g0.y = lds_byte_addr;                                      // lds_addr
  g0.z = (unsigned)ga;                                       // global_addr[31:0]
  g0.w = (unsigned)((ga >> 32) & 0x01ffffffu) | (2u << 30);  // [56:32] | type=2
  tdm_v8i g1;
  g1[0] = (int)(1u << 16);                                   // data_size=1 -> 2B
  g1[1] = (int)((tensor_cols & 0xffffu) << 16);              // tensor_dim0 lo
  g1[2] = (int)((tensor_cols >> 16) | ((tensor_rows & 0xffffu) << 16));
  g1[3] = (int)((tensor_rows >> 16) | (tile_cols << 16));    // tile_dim0
  g1[4] = (int)(tile_rows & 0xffffu);                        // tile_dim1
  g1[5] = (int)row_stride_elems;                             // tensor_dim0_stride
  g1[6] = 0;
  g1[7] = 0;
  tdm_v4i gz4 = {0, 0, 0, 0};                                // 2-D: groups 2/3 off
  tdm_v8i gz8 = {0, 0, 0, 0, 0, 0, 0, 0};
  __builtin_amdgcn_tensor_load_to_lds(g0, g1, gz4, gz4, gz8, 0);
}
#endif

// ---------------- conversion kernels ----------------
__global__ __launch_bounds__(256) void cvt_bf16(const float* __restrict__ in,
                                                __bf16* __restrict__ out, long n) {
  long i = ((long)blockIdx.x * 256 + threadIdx.x) * 4;
  if (i + 3 < n) {
    float4 f = *(const float4*)(in + i);
    out[i + 0] = (__bf16)f.x;
    out[i + 1] = (__bf16)f.y;
    out[i + 2] = (__bf16)f.z;
    out[i + 3] = (__bf16)f.w;
  }
}

// Wt[n*E + k] = (bf16) W[k*E + n]  -> weights become N-major for B fragments
__global__ __launch_bounds__(256) void wtrans(const float* __restrict__ W,
                                              __bf16* __restrict__ Wt) {
  long i = (long)blockIdx.x * 256 + threadIdx.x;
  if (i < (long)Ec * Ec) {
    int k = (int)(i >> 10);
    int n = (int)(i & 1023);
    Wt[(long)n * Ec + k] = (__bf16)W[i];
  }
}

// ---------------- projection GEMM: out[B,H,S,D] = (x @ W + b) * alpha ----------------
__global__ __launch_bounds__(256) void gemm_proj(const __bf16* __restrict__ A,
                                                 const __bf16* __restrict__ Wt,
                                                 const float* __restrict__ bias,
                                                 __bf16* __restrict__ Ohead,
                                                 float alpha) {
#ifdef USE_TDM
  __shared__ alignas(16) __bf16 sA[2][128][32];
  __shared__ alignas(16) __bf16 sB[2][128][32];
#else
  __shared__ alignas(16) __bf16 sA[1][128][32];
  __shared__ alignas(16) __bf16 sB[1][128][32];
#endif
  int tid = threadIdx.x, lane = tid & 31, wave = tid >> 5;
  int bm = blockIdx.x * 128, bn = blockIdx.y * 128;
  int wm = (wave >> 2) * 64, wn = (wave & 3) * 32;

  v8f acc[4][2];
  for (int i = 0; i < 4; ++i)
    for (int j = 0; j < 2; ++j) acc[i][j] = vzero8();

#ifdef USE_TDM
  // TDM double-buffered pipeline: wave 0 drives the DMA, all waves compute.
  const __bf16* Abase = A + (long)bm * Ec;
  const __bf16* Bbase = Wt + (long)bn * Ec;
  if (wave == 0) {
    tdm_load_tile_bf16(Abase, lds_addr_of(&sA[0][0][0]), 32, 128, Ec, Ec, 128);
    tdm_load_tile_bf16(Bbase, lds_addr_of(&sB[0][0][0]), 32, 128, Ec, Ec, 128);
  }
  const int nit = Ec / 32;
  for (int it = 0; it < nit; ++it) {
    int cur = it & 1;
    if (wave == 0) {
      __builtin_amdgcn_s_wait_tensorcnt(0);       // current tiles landed
      if (it + 1 < nit) {                          // prefetch next into other buf
        int k0n = (it + 1) * 32;
        tdm_load_tile_bf16(Abase + k0n, lds_addr_of(&sA[cur ^ 1][0][0]),
                           32, 128, Ec, Ec, 128);
        tdm_load_tile_bf16(Bbase + k0n, lds_addr_of(&sB[cur ^ 1][0][0]),
                           32, 128, Ec, Ec, 128);
      }
    }
    __syncthreads();
    v16bf bfr[2];
#pragma unroll
    for (int nt = 0; nt < 2; ++nt)
      bfr[nt] = load_frag(&sB[cur][wn + nt * 16][0], 32, lane);
#pragma unroll
    for (int mt = 0; mt < 4; ++mt) {
      v16bf a = load_frag(&sA[cur][wm + mt * 16][0], 32, lane);
#pragma unroll
      for (int nt = 0; nt < 2; ++nt) acc[mt][nt] = wmma_bf16(a, bfr[nt], acc[mt][nt]);
    }
    __syncthreads();
  }
#else
  int r  = tid >> 1;
  int cc = (tid & 1) * 16;
  for (int k0 = 0; k0 < Ec; k0 += 32) {
    __syncthreads();
    *(v8bf*)&sA[0][r][cc]     = *(const v8bf*)&A[(long)(bm + r) * Ec + k0 + cc];
    *(v8bf*)&sA[0][r][cc + 8] = *(const v8bf*)&A[(long)(bm + r) * Ec + k0 + cc + 8];
    *(v8bf*)&sB[0][r][cc]     = *(const v8bf*)&Wt[(long)(bn + r) * Ec + k0 + cc];
    *(v8bf*)&sB[0][r][cc + 8] = *(const v8bf*)&Wt[(long)(bn + r) * Ec + k0 + cc + 8];
    __builtin_prefetch(&A[(long)(bm + r) * Ec + k0 + 64], 0, 1);
    __builtin_prefetch(&Wt[(long)(bn + r) * Ec + k0 + 64], 0, 1);
    __syncthreads();
    v16bf bfr[2];
#pragma unroll
    for (int nt = 0; nt < 2; ++nt)
      bfr[nt] = load_frag(&sB[0][wn + nt * 16][0], 32, lane);
#pragma unroll
    for (int mt = 0; mt < 4; ++mt) {
      v16bf a = load_frag(&sA[0][wm + mt * 16][0], 32, lane);
#pragma unroll
      for (int nt = 0; nt < 2; ++nt) acc[mt][nt] = wmma_bf16(a, bfr[nt], acc[mt][nt]);
    }
  }
#endif

#pragma unroll
  for (int mt = 0; mt < 4; ++mt)
#pragma unroll
    for (int nt = 0; nt < 2; ++nt) {
      int col = bn + wn + nt * 16 + (lane & 15);
      int h = col >> 6, d = col & 63;
      float bcol = bias[col];
#pragma unroll
      for (int rr = 0; rr < 8; ++rr) {
        int row = bm + wm + mt * 16 + rr + ((lane >> 4) << 3);
        int b = row >> 12, s = row & (Sc - 1);
        float v = (acc[mt][nt][rr] + bcol) * alpha;
        Ohead[((long)(b * Hc + h) * Sc + s) * Dc + d] = (__bf16)v;
      }
    }
}

// ---------------- windowed + global-column attention ----------------
union WinSmem {
  float  S[16][NCOL];     // fp32 scores (52 KB)
  __bf16 Vt[Dc][128];     // transposed V chunk for P@V
};

__global__ __launch_bounds__(256) void win_attn(const __bf16* __restrict__ qh,
                                                const __bf16* __restrict__ kh,
                                                const __bf16* __restrict__ vh,
                                                __bf16* __restrict__ outh) {
  __shared__ alignas(16) __bf16 sQ[16][Dc];
  __shared__ alignas(16) WinSmem u;
  __shared__ alignas(16) __bf16 sP[16][NCOL];

  int tid = threadIdx.x, lane = tid & 31, wave = tid >> 5;
  int bid = blockIdx.x;
  int qs = bid & 15;
  int c  = (bid >> 4) & 15;
  int h  = (bid >> 8) & 15;
  int b  = bid >> 12;
  long hb = (long)(b * Hc + h) * Sc * Dc;
  const __bf16* qp = qh + hb;
  const __bf16* kp = kh + hb;
  const __bf16* vp = vh + hb;
  int srow0 = c * Wc + qs * 16;

  if (tid < 128) {
    int rr = tid >> 3, cc = (tid & 7) * 8;
    *(v8bf*)&sQ[rr][cc] = *(const v8bf*)&qp[(long)(srow0 + rr) * Dc + cc];
  }
  __syncthreads();

  // ---- phase 1: scores (16 x 832), masked ----
  for (int nt = wave; nt < NCOL / 16; nt += 8) {
    int colb = nt * 16;
    int n = lane & 15;
    long krow;
    if (colb < Gc) krow = colb + n;
    else           krow = iclampr(c * Wc - Wc + (colb - Gc + n));
    v8f acc = vzero8();
#pragma unroll
    for (int kk = 0; kk < Dc; kk += 32) {
      v16bf a = load_frag(&sQ[0][kk], Dc, lane);
      int kb = ((lane >> 4) << 3) + kk;
      const __bf16* p = kp + krow * Dc + kb;
      FragU f;
      f.h[0] = *(const v8bf*)p;
      f.h[1] = *(const v8bf*)(p + 16);
      acc = wmma_bf16(a, f.v, acc);
    }
    int ncol = colb + n;
#pragma unroll
    for (int rr = 0; rr < 8; ++rr) {
      int m = rr + ((lane >> 4) << 3);
      float val = acc[rr];
      if (ncol >= Gc) {
        int j = ncol - Gc;
        int i = qs * 16 + m;
        int kpos = c * Wc - Wc + j;
        bool ok = (j >= i) && (j <= i + 2 * Wc) && (kpos >= Gc) && (kpos < Sc);
        if (!ok) val = -1e9f;
      }
      u.S[m][ncol] = val;
    }
  }
  __syncthreads();

  // ---- phase 2: softmax over 832, 16 lanes per row ----
  {
    int row = tid >> 4;
    int sub = tid & 15;
    float mx = -3.4e38f;
    for (int j = sub; j < NCOL; j += 16) mx = fmaxf(mx, u.S[row][j]);
    for (int o = 8; o > 0; o >>= 1) mx = fmaxf(mx, __shfl_xor(mx, o, 32));
    float sum = 0.f;
    for (int j = sub; j < NCOL; j += 16) {
      float e = __expf(u.S[row][j] - mx);
      sum += e;
      u.S[row][j] = e;
    }
    for (int o = 8; o > 0; o >>= 1) sum += __shfl_xor(sum, o, 32);
    float inv = 1.f / sum;
    for (int j = sub; j < NCOL; j += 16) sP[row][j] = (__bf16)(u.S[row][j] * inv);
  }

  // ---- phase 3: out(16x64) = P(16x832) @ V(832x64), V chunk-transposed in LDS ----
  v8f oacc = vzero8();
  for (int k0 = 0; k0 < NCOL; k0 += 128) {
    int ck = imin(128, NCOL - k0);
    __syncthreads();
    for (int t = tid; t < ck * 8; t += 256) {
      int key = k0 + (t >> 3);
      int dh = (t & 7) * 8;
      long krow = (key < Gc) ? key : (long)iclampr(c * Wc - Wc + (key - Gc));
      v8bf vv = *(const v8bf*)&vp[krow * Dc + dh];
#pragma unroll
      for (int e = 0; e < 8; ++e) u.Vt[dh + e][t >> 3] = vv[e];
    }
    __syncthreads();
    if (wave < 4) {
      for (int kk = 0; kk < ck; kk += 32) {
        v16bf a  = load_frag(&sP[0][k0 + kk], NCOL, lane);
        v16bf bb = load_frag(&u.Vt[wave * 16][kk], 128, lane);
        oacc = wmma_bf16(a, bb, oacc);
      }
    }
  }
  if (wave < 4) {
    int n = wave * 16 + (lane & 15);
#pragma unroll
    for (int rr = 0; rr < 8; ++rr) {
      int m = rr + ((lane >> 4) << 3);
      outh[hb + (long)(srow0 + m) * Dc + n] = (__bf16)oacc[rr];
    }
  }
}

// ---------------- global-row attention (first G rows) ----------------
__global__ __launch_bounds__(256) void g_scores(const __bf16* __restrict__ qgh,
                                                const __bf16* __restrict__ kgh,
                                                float* __restrict__ sg) {
  __shared__ alignas(16) __bf16 sQ[Gc][Dc];
  int bh = blockIdx.x;
  int n128 = blockIdx.y * 128;
  const __bf16* qp = qgh + (long)bh * Sc * Dc;
  const __bf16* kp = kgh + (long)bh * Sc * Dc;
  float* sp = sg + (long)bh * Gc * Sc;
  int tid = threadIdx.x, lane = tid & 31, wave = tid >> 5;
  {
    int rr = tid >> 2, cc = (tid & 3) * 16;
    *(v8bf*)&sQ[rr][cc]     = *(const v8bf*)&qp[(long)rr * Dc + cc];
    *(v8bf*)&sQ[rr][cc + 8] = *(const v8bf*)&qp[(long)rr * Dc + cc + 8];
  }
  __syncthreads();
  int nt0 = n128 + wave * 16;
#pragma unroll
  for (int mt = 0; mt < 4; ++mt) {
    v8f acc = vzero8();
#pragma unroll
    for (int kk = 0; kk < Dc; kk += 32) {
      v16bf a  = load_frag(&sQ[mt * 16][kk], Dc, lane);
      v16bf bb = load_frag(&kp[(long)nt0 * Dc + kk], Dc, lane);
      acc = wmma_bf16(a, bb, acc);
    }
    int n = nt0 + (lane & 15);
#pragma unroll
    for (int rr = 0; rr < 8; ++rr) {
      int m = mt * 16 + rr + ((lane >> 4) << 3);
      sp[(long)m * Sc + n] = acc[rr];
    }
  }
}

__global__ __launch_bounds__(256) void g_softmax(const float* __restrict__ sg,
                                                 __bf16* __restrict__ pg) {
  __shared__ float red[8], red2[8];
  long row = blockIdx.x;
  const float* sp = sg + row * Sc;
  __bf16* pp = pg + row * Sc;
  int tid = threadIdx.x;
  float mx = -3.4e38f;
  for (int j = tid; j < Sc; j += 256) mx = fmaxf(mx, sp[j]);
  for (int o = 16; o > 0; o >>= 1) mx = fmaxf(mx, __shfl_xor(mx, o, 32));
  if ((tid & 31) == 0) red[tid >> 5] = mx;
  __syncthreads();
  mx = red[0];
  for (int i = 1; i < 8; ++i) mx = fmaxf(mx, red[i]);
  float sum = 0.f;
  for (int j = tid; j < Sc; j += 256) sum += __expf(sp[j] - mx);
  for (int o = 16; o > 0; o >>= 1) sum += __shfl_xor(sum, o, 32);
  if ((tid & 31) == 0) red2[tid >> 5] = sum;
  __syncthreads();
  sum = 0.f;
  for (int i = 0; i < 8; ++i) sum += red2[i];
  float inv = 1.f / sum;
  for (int j = tid; j < Sc; j += 256) pp[j] = (__bf16)(__expf(sp[j] - mx) * inv);
}

__global__ __launch_bounds__(256) void g_out(const __bf16* __restrict__ pg,
                                             const __bf16* __restrict__ vgh,
                                             __bf16* __restrict__ outh) {
  __shared__ alignas(16) __bf16 sVt[Dc][128];
  int bh = blockIdx.x;
  const __bf16* pp = pg + (long)bh * Gc * Sc;
  const __bf16* vp = vgh + (long)bh * Sc * Dc;
  int tid = threadIdx.x, lane = tid & 31, wave = tid >> 5;
  int mt  = (wave >> 1) * 16;
  int ntA = (wave & 1) * 32;
  v8f acc[2] = {vzero8(), vzero8()};
  for (int k0 = 0; k0 < Sc; k0 += 128) {
    __syncthreads();
    for (int t = tid; t < 128 * 8; t += 256) {
      int key = t >> 3, dh = (t & 7) * 8;
      v8bf vv = *(const v8bf*)&vp[(long)(k0 + key) * Dc + dh];
#pragma unroll
      for (int e = 0; e < 8; ++e) sVt[dh + e][key] = vv[e];
    }
    __syncthreads();
#pragma unroll
    for (int kk = 0; kk < 128; kk += 32) {
      v16bf a = load_frag(&pp[(long)mt * Sc + k0 + kk], Sc, lane);
#pragma unroll
      for (int t2 = 0; t2 < 2; ++t2) {
        v16bf bb = load_frag(&sVt[ntA + t2 * 16][kk], 128, lane);
        acc[t2] = wmma_bf16(a, bb, acc[t2]);
      }
    }
  }
  long obase = (long)bh * Sc * Dc;
#pragma unroll
  for (int t2 = 0; t2 < 2; ++t2) {
    int n = ntA + t2 * 16 + (lane & 15);
#pragma unroll
    for (int rr = 0; rr < 8; ++rr) {
      int m = mt + rr + ((lane >> 4) << 3);
      outh[obase + (long)m * Dc + n] = (__bf16)acc[t2][rr];
    }
  }
}

// ---------------- output GEMM: d_out = ctx @ Wo + bo (ctx gathered from head layout) ---
__global__ __launch_bounds__(256) void gemm_out(const __bf16* __restrict__ Ahead,
                                                const __bf16* __restrict__ Wt,
                                                const float* __restrict__ bias,
                                                float* __restrict__ Out) {
#ifdef USE_TDM
  __shared__ alignas(16) __bf16 sA[2][128][32];
  __shared__ alignas(16) __bf16 sB[2][128][32];
#else
  __shared__ alignas(16) __bf16 sA[1][128][32];
  __shared__ alignas(16) __bf16 sB[1][128][32];
#endif
  int tid = threadIdx.x, lane = tid & 31, wave = tid >> 5;
  int bm = blockIdx.x * 128, bn = blockIdx.y * 128;
  int wm = (wave >> 2) * 64, wn = (wave & 3) * 32;

  v8f acc[4][2];
  for (int i = 0; i < 4; ++i)
    for (int j = 0; j < 2; ++j) acc[i][j] = vzero8();

#ifdef USE_TDM
  int b0 = bm >> 12, s0 = bm & (Sc - 1);
  const __bf16* Bbase = Wt + (long)bn * Ec;
  auto a_tile = [&](int k0) -> const __bf16* {
    int hh = k0 >> 6, doff = k0 & 63;
    return Ahead + ((long)(b0 * Hc + hh) * Sc + s0) * Dc + doff;
  };
  if (wave == 0) {
    tdm_load_tile_bf16(a_tile(0), lds_addr_of(&sA[0][0][0]), 32, 128, Dc, Dc, 128);
    tdm_load_tile_bf16(Bbase, lds_addr_of(&sB[0][0][0]), 32, 128, Ec, Ec, 128);
  }
  const int nit = Ec / 32;
  for (int it = 0; it < nit; ++it) {
    int cur = it & 1;
    if (wave == 0) {
      __builtin_amdgcn_s_wait_tensorcnt(0);
      if (it + 1 < nit) {
        int k0n = (it + 1) * 32;
        tdm_load_tile_bf16(a_tile(k0n), lds_addr_of(&sA[cur ^ 1][0][0]),
                           32, 128, Dc, Dc, 128);
        tdm_load_tile_bf16(Bbase + k0n, lds_addr_of(&sB[cur ^ 1][0][0]),
                           32, 128, Ec, Ec, 128);
      }
    }
    __syncthreads();
    v16bf bfr[2];
#pragma unroll
    for (int nt = 0; nt < 2; ++nt)
      bfr[nt] = load_frag(&sB[cur][wn + nt * 16][0], 32, lane);
#pragma unroll
    for (int mt = 0; mt < 4; ++mt) {
      v16bf a = load_frag(&sA[cur][wm + mt * 16][0], 32, lane);
#pragma unroll
      for (int nt = 0; nt < 2; ++nt) acc[mt][nt] = wmma_bf16(a, bfr[nt], acc[mt][nt]);
    }
    __syncthreads();
  }
#else
  int r  = tid >> 1;
  int cc = (tid & 1) * 16;
  int row = bm + r;
  int bb_ = row >> 12, s = row & (Sc - 1);
  for (int k0 = 0; k0 < Ec; k0 += 32) {
    __syncthreads();
    int hh = k0 >> 6, doff = k0 & 63;
    const __bf16* ap = Ahead + ((long)(bb_ * Hc + hh) * Sc + s) * Dc + doff;
    *(v8bf*)&sA[0][r][cc]     = *(const v8bf*)&ap[cc];
    *(v8bf*)&sA[0][r][cc + 8] = *(const v8bf*)&ap[cc + 8];
    *(v8bf*)&sB[0][r][cc]     = *(const v8bf*)&Wt[(long)(bn + r) * Ec + k0 + cc];
    *(v8bf*)&sB[0][r][cc + 8] = *(const v8bf*)&Wt[(long)(bn + r) * Ec + k0 + cc + 8];
    __builtin_prefetch(&Wt[(long)(bn + r) * Ec + k0 + 64], 0, 1);
    __syncthreads();
    v16bf bfr[2];
#pragma unroll
    for (int nt = 0; nt < 2; ++nt)
      bfr[nt] = load_frag(&sB[0][wn + nt * 16][0], 32, lane);
#pragma unroll
    for (int mt = 0; mt < 4; ++mt) {
      v16bf a = load_frag(&sA[0][wm + mt * 16][0], 32, lane);
#pragma unroll
      for (int nt = 0; nt < 2; ++nt) acc[mt][nt] = wmma_bf16(a, bfr[nt], acc[mt][nt]);
    }
  }
#endif

#pragma unroll
  for (int mt = 0; mt < 4; ++mt)
#pragma unroll
    for (int nt = 0; nt < 2; ++nt) {
      int col = bn + wn + nt * 16 + (lane & 15);
      float bcol = bias[col];
#pragma unroll
      for (int rr = 0; rr < 8; ++rr) {
        int orow = bm + wm + mt * 16 + rr + ((lane >> 4) << 3);
        Out[(long)orow * Ec + col] = acc[mt][nt][rr] + bcol;
      }
    }
}

// ---------------- host launcher ----------------
extern "C" void kernel_launch(void* const* d_in, const int* in_sizes, int n_in,
                              void* d_out, int out_size, void* d_ws, size_t ws_size,
                              hipStream_t stream) {
  (void)in_sizes; (void)n_in; (void)out_size; (void)ws_size;
  const float* x   = (const float*)d_in[0];
  const float* Wq  = (const float*)d_in[2];  const float* bq  = (const float*)d_in[3];
  const float* Wk  = (const float*)d_in[4];  const float* bk  = (const float*)d_in[5];
  const float* Wv  = (const float*)d_in[6];  const float* bv  = (const float*)d_in[7];
  const float* Wqg = (const float*)d_in[8];  const float* bqg = (const float*)d_in[9];
  const float* Wkg = (const float*)d_in[10]; const float* bkg = (const float*)d_in[11];
  const float* Wvg = (const float*)d_in[12]; const float* bvg = (const float*)d_in[13];
  const float* Wo  = (const float*)d_in[14]; const float* bo  = (const float*)d_in[15];

  char* ws = (char*)d_ws;
  size_t off = 0;
  auto alloc = [&](size_t bytes) -> void* {
    void* p = ws + off;
    off += (bytes + 255) & ~(size_t)255;
    return p;
  };
  const size_t actN = (size_t)BSr * Ec;            // 8192*1024 elements
  __bf16* xb = (__bf16*)alloc(actN * 2);
  __bf16* WtA[7];
  for (int i = 0; i < 7; ++i) WtA[i] = (__bf16*)alloc((size_t)Ec * Ec * 2);
  __bf16* qhb  = (__bf16*)alloc(actN * 2);
  __bf16* khb  = (__bf16*)alloc(actN * 2);
  __bf16* vhb  = (__bf16*)alloc(actN * 2);
  __bf16* qghb = (__bf16*)alloc(actN * 2);
  __bf16* kghb = (__bf16*)alloc(actN * 2);
  __bf16* vghb = (__bf16*)alloc(actN * 2);
  __bf16* outh = (__bf16*)alloc(actN * 2);
  float*  sg   = (float*) alloc((size_t)Bc * Hc * Gc * Sc * 4);
  __bf16* pg   = (__bf16*)alloc((size_t)Bc * Hc * Gc * Sc * 2);

  const float scale = 0.125f;  // 1/sqrt(64)

  cvt_bf16<<<(int)(actN / (256 * 4)), 256, 0, stream>>>(x, xb, (long)actN);
  const float* Wsrc[7] = {Wq, Wk, Wv, Wqg, Wkg, Wvg, Wo};
  for (int i = 0; i < 7; ++i)
    wtrans<<<(Ec * Ec) / 256, 256, 0, stream>>>(Wsrc[i], WtA[i]);

  dim3 gG(64, 8);  // 8192/128 x 1024/128
  gemm_proj<<<gG, 256, 0, stream>>>(xb, WtA[0], bq,  qhb,  scale);
  gemm_proj<<<gG, 256, 0, stream>>>(xb, WtA[1], bk,  khb,  1.f);
  gemm_proj<<<gG, 256, 0, stream>>>(xb, WtA[2], bv,  vhb,  1.f);
  gemm_proj<<<gG, 256, 0, stream>>>(xb, WtA[3], bqg, qghb, scale);
  gemm_proj<<<gG, 256, 0, stream>>>(xb, WtA[4], bkg, kghb, 1.f);
  gemm_proj<<<gG, 256, 0, stream>>>(xb, WtA[5], bvg, vghb, 1.f);

  win_attn<<<Bc * Hc * NCc * 16, 256, 0, stream>>>(qhb, khb, vhb, outh);

  g_scores<<<dim3(Bc * Hc, Sc / 128), 256, 0, stream>>>(qghb, kghb, sg);
  g_softmax<<<Bc * Hc * Gc, 256, 0, stream>>>(sg, pg);
  g_out<<<Bc * Hc, 256, 0, stream>>>(pg, vghb, outh);

  gemm_out<<<gG, 256, 0, stream>>>(outh, WtA[6], bo, (float*)d_out);
}
